// TransformerAttention_21818433863681
// MI455X (gfx1250) — matrix operand order repaired and checked by reading, not compiled, not run on hardware
//
#include <hip/hip_runtime.h>
#include <hip/hip_bf16.h>

// ---------------------------------------------------------------------------
// CDNA5 (gfx1250) GQA attention block, bf16 WMMA (v_wmma_f32_16x16x32_bf16),
// f32 accumulation, wave32. Flash kernel stages V tiles into LDS with
// global_load_async_to_lds_b128 (ASYNCcnt) double-buffered against compute.
// ---------------------------------------------------------------------------

typedef __bf16 bf16_t;
typedef __attribute__((ext_vector_type(16))) __bf16 v16bf;
typedef __attribute__((ext_vector_type(8)))  __bf16 v8bf;
typedef __attribute__((ext_vector_type(8)))  float  v8f;

static __device__ inline v8f v8f_zero() {
  v8f v;
#pragma unroll
  for (int i = 0; i < 8; ++i) v[i] = 0.0f;
  return v;
}

// A-fragment (16x32 bf16, M x K) per ISA 7.12.2:
//   lane m (both halves M = lane%16); lane<16 holds K {0..7, 16..23},
//   lane>=16 holds K {8..15, 24..31}.  Two 16B loads per lane.
static __device__ inline v16bf load_a_frag(const bf16_t* __restrict__ base, int ld) {
  const int lane = threadIdx.x & 31;
  const int row  = lane & 15;
  const int c0   = (lane >> 4) << 3;            // 0 or 8
  const bf16_t* p = base + (size_t)row * ld + c0;
  v8bf lo = *reinterpret_cast<const v8bf*>(p);
  v8bf hi = *reinterpret_cast<const v8bf*>(p + 16);
  v16bf a;
#pragma unroll
  for (int i = 0; i < 8; ++i) { a[i] = lo[i]; a[8 + i] = hi[i]; }
  return a;
}

// B-fragment (32x16 bf16, K x N): lane n = lane%16; lane<16 holds K {0..7, 16..23},
// lane>=16 holds K {8..15, 24..31}.  Strided scalar loads.
static __device__ inline v16bf load_b_frag(const bf16_t* __restrict__ base, int ld) {
  const int lane = threadIdx.x & 31;
  const int col  = lane & 15;
  const int k0   = (lane >> 4) << 3;
  const bf16_t* p = base + (size_t)k0 * ld + col;
  v16bf b;
#pragma unroll
  for (int i = 0; i < 8; ++i) { b[i] = p[(size_t)i * ld]; b[8 + i] = p[(size_t)(16 + i) * ld]; }
  return b;
}

static __device__ inline v8f wmma_bf16(v16bf a, v16bf b, v8f c) {
  // (neg_a, A, neg_b, B, c_mod, C, reuse_a, reuse_b)
  return __builtin_amdgcn_wmma_f32_16x16x32_bf16(false, a, false, b, (short)0, c,
                                                 false, false);
}

// ---- CDNA5 async copy: global -> LDS, tracked by ASYNCcnt ------------------
// Copies a 32x64-element bf16 tile (4KB) into LDS with padded row stride of
// 72 elements (144B) for conflict-free strided 16-bit B-fragment reads.
// 8 instructions x (32 lanes x 16B) = 4KB.
#define VT_STRIDE 72  // elements; 144 bytes per row

static __device__ inline void async_copy_v_tile(const bf16_t* __restrict__ g,
                                                uint32_t lds_base) {
  const int lane = threadIdx.x & 31;
  const int rowi = lane >> 3;               // 0..3
  const int colb = (lane & 7) * 16;         // byte offset within a 128B row
#pragma unroll
  for (int i = 0; i < 8; ++i) {
    const int row = i * 4 + rowi;           // 0..31
    const uint64_t ga = (uint64_t)(uintptr_t)g + (uint64_t)row * 128 + colb;
    const uint32_t la = lds_base + (uint32_t)row * (VT_STRIDE * 2) + colb;
    asm volatile("global_load_async_to_lds_b128 %0, %1, off"
                 :: "v"(la), "v"(ga)
                 : "memory");
  }
}

static __device__ inline void wait_async_done() {
  asm volatile("s_wait_asynccnt 0" ::: "memory");
}

// ---------------------------------------------------------------------------
// Precompute rope factor rf[pos, d] = cos(ang) + sign(pos)*sin(ang),
// sign = -1 for even positions.  theta = 10000^(-2*(d/2)/64).
// ---------------------------------------------------------------------------
__global__ __launch_bounds__(256) void rope_kernel(float* __restrict__ rf) {
  const int t = blockIdx.x * 256 + threadIdx.x;
  if (t >= 2048 * 64) return;
  const int pos = t >> 6;
  const int i   = t & 63;
  const float theta = __expf(-(2.0f * (float)(i >> 1) / 64.0f) * 9.210340371976184f);
  const float ang   = (float)pos * theta;
  const float sgn   = (pos & 1) ? 1.0f : -1.0f;
  rf[t] = __cosf(ang) + sgn * __sinf(ang);
}

// ---------------------------------------------------------------------------
// RMSNorm per row of 1024, fp32 in, bf16 out (scaled by norm_w).
// ---------------------------------------------------------------------------
__global__ __launch_bounds__(256) void rmsnorm_kernel(const float* __restrict__ x,
                                                      const float* __restrict__ w,
                                                      bf16_t* __restrict__ xn) {
  __shared__ float red[256];
  const int row = blockIdx.x;                      // 0..4095
  const float* xr = x + (size_t)row * 1024;
  float ss = 0.0f;
  for (int i = threadIdx.x; i < 1024; i += 256) {
    const float v = xr[i];
    ss += v * v;
  }
  red[threadIdx.x] = ss;
  __syncthreads();
  for (int s = 128; s > 0; s >>= 1) {
    if ((int)threadIdx.x < s) red[threadIdx.x] += red[threadIdx.x + s];
    __syncthreads();
  }
  const float inv = rsqrtf(red[0] * (1.0f / 1024.0f) + 1.1920929e-07f);
  for (int i = threadIdx.x; i < 1024; i += 256)
    xn[(size_t)row * 1024 + i] = (bf16_t)(xr[i] * inv * w[i]);
}

__global__ __launch_bounds__(256) void cast_bf16_kernel(const float* __restrict__ in,
                                                        bf16_t* __restrict__ out, int n) {
  for (int i = blockIdx.x * 256 + threadIdx.x; i < n; i += gridDim.x * 256)
    out[i] = (bf16_t)in[i];
}

// ---------------------------------------------------------------------------
// Generic bf16 GEMM: one wave -> 16x128 tile of C = A[4096,K] @ W[K,N].
// A fragment reused across 8 B fragments per K-chunk (0.25 b128 A loads + 16 scalar B loads / WMMA).
// Epilogues: 0=V store [bk,key,64]; 1=Q (*rope) [m,1024];
//            2=K (*rope) transposed [bk*64+d, pos]; 3=f32 residual (x + C).
// ---------------------------------------------------------------------------
__global__ __launch_bounds__(32) void gemm16_kernel(
    const bf16_t* __restrict__ A, const bf16_t* __restrict__ W,
    int K, int N, int ep,
    const float* __restrict__ rope, const float* __restrict__ xres,
    bf16_t* __restrict__ outb, float* __restrict__ outf) {
  const int lane  = threadIdx.x & 31;
  const int mtile = blockIdx.x;
  const int nbase = blockIdx.y * 128;

  v8f acc[8];
#pragma unroll
  for (int t = 0; t < 8; ++t) acc[t] = v8f_zero();

  const bf16_t* arow = A + (size_t)mtile * 16 * K;
  for (int k = 0; k < K; k += 32) {
    const v16bf a = load_a_frag(arow + k, K);
#pragma unroll
    for (int t = 0; t < 8; ++t) {
      const v16bf b = load_b_frag(W + (size_t)k * N + nbase + t * 16, N);
      acc[t] = wmma_bf16(a, b, acc[t]);
    }
  }

  const int rowSel = (lane >> 4) << 3;   // 0 or 8 (C layout: lanes>=16 -> M+8)
  const int nlane  = lane & 15;
#pragma unroll
  for (int t = 0; t < 8; ++t) {
#pragma unroll
    for (int r = 0; r < 8; ++r) {
      float val = acc[t][r];
      const int    M   = r + rowSel;
      const size_t m   = (size_t)mtile * 16 + M;   // global row (b*2048+pos)
      const int    nc  = nbase + t * 16 + nlane;   // output column
      const int    bq  = (int)(m >> 11);
      const int    pos = (int)(m & 2047);
      if (ep == 1) {                                // Q
        val *= rope[pos * 64 + (nc & 63)];
        outb[m * (size_t)N + nc] = (bf16_t)val;
      } else if (ep == 2) {                         // K -> Kt[(bq*4+kvh)*64+d][pos]
        const int kvh = nc >> 6, d = nc & 63;
        val *= rope[pos * 64 + d];
        outb[((size_t)(bq * 4 + kvh) * 64 + d) * 2048 + pos] = (bf16_t)val;
      } else if (ep == 0) {                         // V -> [(bq*4+kvh)*2048+key][d]
        const int kvh = nc >> 6, d = nc & 63;
        outb[((size_t)(bq * 4 + kvh) * 2048 + pos) * 64 + d] = (bf16_t)val;
      } else {                                      // O: residual, fp32
        outf[m * (size_t)N + nc] = xres[m * (size_t)N + nc] + val;
      }
    }
  }
}

// ---------------------------------------------------------------------------
// Flash attention: one wave per (batch, head, 16-query tile).
// Streams causal 32-key tiles: 4 score WMMA + online softmax + 4 PV WMMA.
// V tiles are staged into double-buffered LDS by global_load_async_to_lds_b128
// (issued one tile ahead, overlapping the PV + next score WMMAs); P (16x32)
// is bounced through LDS to re-layout C-frag -> A-frag (bf16).
// ---------------------------------------------------------------------------
__global__ __launch_bounds__(32) void flash_kernel(
    const bf16_t* __restrict__ Qb,   // [4096, 1024]      (rope applied)
    const bf16_t* __restrict__ Ktb,  // [(b*4+kvh)*64+d, 2048] (rope applied)
    const bf16_t* __restrict__ Vb,   // [(b*4+kvh)*2048+key, 64]
    bf16_t* __restrict__ attnb) {    // [4096, 1024]
  __shared__ __align__(16) bf16_t pbuf[16 * 32];
  __shared__ __align__(16) bf16_t vbuf[2][32 * VT_STRIDE];
  const int lane  = threadIdx.x & 31;
  const int qt    = blockIdx.x;      // 0..127
  const int h     = blockIdx.y;      // 0..15
  const int bq    = blockIdx.z;      // 0..1
  const int kvh   = h & 3;           // torch repeat: kv head = h % KV_HEADS
  const int bk    = bq * 4 + kvh;
  const int qpos0 = qt * 16;

  const bf16_t* qbase = Qb + ((size_t)(bq * 2048 + qpos0)) * 1024 + h * 64;
  const v16bf qa0 = load_a_frag(qbase, 1024);        // d = 0..31
  const v16bf qa1 = load_a_frag(qbase + 32, 1024);   // d = 32..63

  v8f O[4];
  float rM[8], rL[8];
#pragma unroll
  for (int t = 0; t < 4; ++t) O[t] = v8f_zero();
#pragma unroll
  for (int r = 0; r < 8; ++r) { rM[r] = -1e30f; rL[r] = 0.0f; }

  const int rowSel = (lane >> 4) << 3;
  const int nlane  = lane & 15;
  const int jt_end = (qpos0 + 15) >> 5;              // inclusive, causal bound

  const bf16_t* vtile0 = Vb + (size_t)bk * 2048 * 64;
  const uint32_t vb_lds[2] = { (uint32_t)(uintptr_t)&vbuf[0][0],
                               (uint32_t)(uintptr_t)&vbuf[1][0] };
  // Prefetch first V tile (keys 0..31) into buffer 0.
  async_copy_v_tile(vtile0, vb_lds[0]);

  for (int jt = 0; jt <= jt_end; ++jt) {
    const int kpos0 = jt << 5;
    v8f s0 = v8f_zero(), s1 = v8f_zero();
    const bf16_t* kb0 = Ktb + ((size_t)bk * 64 + 0)  * 2048 + kpos0;
    const bf16_t* kb1 = Ktb + ((size_t)bk * 64 + 32) * 2048 + kpos0;
    s0 = wmma_bf16(qa0, load_b_frag(kb0,      2048), s0);
    s0 = wmma_bf16(qa1, load_b_frag(kb1,      2048), s0);
    s1 = wmma_bf16(qa0, load_b_frag(kb0 + 16, 2048), s1);
    s1 = wmma_bf16(qa1, load_b_frag(kb1 + 16, 2048), s1);

    float corr[8];
#pragma unroll
    for (int r = 0; r < 8; ++r) {
      const int ig = qpos0 + r + rowSel;             // global query position
      const int j0 = kpos0 + nlane;                  // key positions j0, j0+16
      float a0 = (j0      <= ig) ? s0[r] * 0.125f : -1e30f;
      float a1 = (j0 + 16 <= ig) ? s1[r] * 0.125f : -1e30f;
      float mx = fmaxf(a0, a1);
#pragma unroll
      for (int off = 1; off < 16; off <<= 1)
        mx = fmaxf(mx, __shfl_xor(mx, off, 32));     // row-max within half
      const float mn = fmaxf(rM[r], mx);
      const float c  = __expf(rM[r] - mn);
      corr[r] = c;
      const float p0 = __expf(a0 - mn);
      const float p1 = __expf(a1 - mn);
      float rs = p0 + p1;
#pragma unroll
      for (int off = 1; off < 16; off <<= 1)
        rs += __shfl_xor(rs, off, 32);               // row-sum within half
      rL[r] = rL[r] * c + rs;
      rM[r] = mn;
      // C-layout -> natural [row][col] bf16 in LDS
      pbuf[(r + rowSel) * 32 + nlane]      = (bf16_t)p0;
      pbuf[(r + rowSel) * 32 + 16 + nlane] = (bf16_t)p1;
    }
#pragma unroll
    for (int t = 0; t < 4; ++t)
#pragma unroll
      for (int r = 0; r < 8; ++r) O[t][r] *= corr[r];

    // V(jt) must be resident; then immediately prefetch V(jt+1) into the
    // other buffer so the copy overlaps PV + next tile's score WMMAs.
    wait_async_done();
    if (jt + 1 <= jt_end)
      async_copy_v_tile(vtile0 + (size_t)(kpos0 + 32) * 64, vb_lds[(jt + 1) & 1]);

    // P as A-fragment (LDS, DS ops are in-order within the wave)
    const v16bf pa = load_a_frag(pbuf, 32);
    const bf16_t* vcur = &vbuf[jt & 1][0];
#pragma unroll
    for (int t = 0; t < 4; ++t)
      O[t] = wmma_bf16(pa, load_b_frag(vcur + t * 16, VT_STRIDE), O[t]);
  }

#pragma unroll
  for (int t = 0; t < 4; ++t)
#pragma unroll
    for (int r = 0; r < 8; ++r) {
      const int row = r + rowSel;
      const float val = O[t][r] / rL[r];
      attnb[((size_t)(bq * 2048 + qpos0 + row)) * 1024 + h * 64 + t * 16 + nlane] =
          (bf16_t)val;
    }
}

// ---------------------------------------------------------------------------
extern "C" void kernel_launch(void* const* d_in, const int* in_sizes, int n_in,
                              void* d_out, int out_size, void* d_ws, size_t ws_size,
                              hipStream_t stream) {
  (void)in_sizes; (void)n_in; (void)out_size; (void)ws_size;
  const float* x      = (const float*)d_in[0];   // [2,2048,1024]
  const float* w_q    = (const float*)d_in[1];   // [1024,1024]
  const float* w_k    = (const float*)d_in[2];   // [1024,256]
  const float* w_v    = (const float*)d_in[3];   // [1024,256]
  const float* w_o    = (const float*)d_in[4];   // [1024,1024]
  const float* norm_w = (const float*)d_in[5];   // [1024]
  float* out = (float*)d_out;                    // [2,2048,1024]

  char* wsp = (char*)d_ws;
  auto take = [&](size_t bytes) -> void* {
    void* p = (void*)wsp;
    wsp += (bytes + 255) & ~(size_t)255;
    return p;
  };
  bf16_t* xn   = (bf16_t*)take(4096ull * 1024 * 2);
  bf16_t* wqb  = (bf16_t*)take(1024ull * 1024 * 2);
  bf16_t* wkb  = (bf16_t*)take(1024ull * 256 * 2);
  bf16_t* wvb  = (bf16_t*)take(1024ull * 256 * 2);
  bf16_t* wob  = (bf16_t*)take(1024ull * 1024 * 2);
  bf16_t* Qb   = (bf16_t*)take(4096ull * 1024 * 2);
  bf16_t* Ktb  = (bf16_t*)take(8ull * 64 * 2048 * 2);
  bf16_t* Vb   = (bf16_t*)take(8ull * 2048 * 64 * 2);
  bf16_t* Attn = (bf16_t*)take(4096ull * 1024 * 2);
  float*  rope = (float*) take(2048ull * 64 * 4);

  rope_kernel<<<512, 256, 0, stream>>>(rope);
  rmsnorm_kernel<<<4096, 256, 0, stream>>>(x, norm_w, xn);
  cast_bf16_kernel<<<1024, 256, 0, stream>>>(w_q, wqb, 1024 * 1024);
  cast_bf16_kernel<<<256,  256, 0, stream>>>(w_k, wkb, 1024 * 256);
  cast_bf16_kernel<<<256,  256, 0, stream>>>(w_v, wvb, 1024 * 256);
  cast_bf16_kernel<<<1024, 256, 0, stream>>>(w_o, wob, 1024 * 1024);

  // Q / K / V projections (rope fused into Q and K epilogues)
  gemm16_kernel<<<dim3(256, 8), 32, 0, stream>>>(xn, wqb, 1024, 1024, 1, rope,
                                                 nullptr, Qb, nullptr);
  gemm16_kernel<<<dim3(256, 2), 32, 0, stream>>>(xn, wkb, 1024, 256, 2, rope,
                                                 nullptr, Ktb, nullptr);
  gemm16_kernel<<<dim3(256, 2), 32, 0, stream>>>(xn, wvb, 1024, 256, 0, rope,
                                                 nullptr, Vb, nullptr);

  // Causal GQA flash attention
  flash_kernel<<<dim3(128, 16, 2), 32, 0, stream>>>(Qb, Ktb, Vb, Attn);

  // Output projection + residual
  gemm16_kernel<<<dim3(256, 8), 32, 0, stream>>>(Attn, wob, 1024, 1024, 3, rope,
                                                 x, nullptr, out);
}
